// MultiHeadAttention_50792283242733
// MI455X (gfx1250) — matrix operand order repaired and checked by reading, not compiled, not run on hardware
//
#include <hip/hip_runtime.h>
#include <hip/hip_bf16.h>
#include <math.h>

// ---------------------------------------------------------------------------
// MI455X (gfx1250) multi-head attention, flash-style, f16 WMMA w/ f32 accum.
// B=2, S=2048, E=1024, H=16, D=64.
// ---------------------------------------------------------------------------

typedef _Float16     v16h  __attribute__((ext_vector_type(16)));
typedef _Float16     v4h   __attribute__((ext_vector_type(4)));
typedef float        v8f   __attribute__((ext_vector_type(8)));
typedef float        v4f   __attribute__((ext_vector_type(4)));
typedef unsigned int u32v4 __attribute__((ext_vector_type(4)));

#define BATCH 2
#define SEQ   2048
#define EMB   1024
#define NH    16
#define HD    64

union FragU { u32v4 u[2]; v16h h; };

#define WMMA_F16(a, b, c) \
  __builtin_amdgcn_wmma_f32_16x16x32_f16(false, (a), false, (b), (short)0, (c), false, false)

// A-fragment (16x32 f16, M x K): lane L holds row M = L&15.
// halves 0..7  -> K = k0 + 8*(L>>4)   .. +7    (contiguous, 16B)
// halves 8..15 -> K = k0 + 16 + 8*(L>>4) .. +7 (contiguous, 16B)
static __device__ __forceinline__ v16h load_a_frag(const _Float16* base, int ldk,
                                                   int m0, int k0) {
  const int lane = threadIdx.x & 31;
  const int lo = lane & 15, hi = lane >> 4;
  const _Float16* p = base + (size_t)(m0 + lo) * ldk;
  FragU f;
  f.u[0] = *(const u32v4*)(p + k0 + 8 * hi);
  f.u[1] = *(const u32v4*)(p + k0 + 16 + 8 * hi);
  return f.h;
}

// B-fragment (32x16 f16, K x N) loaded from B^T stored row-major [N][K]:
// lane L holds column N = L&15; halves 0..7 -> K = k0 + 8*(L>>4) .. +7,
// halves 8..15 -> K = k0 + 16 + 8*(L>>4) .. +7.
static __device__ __forceinline__ v16h load_b_frag(const _Float16* base, int ldk,
                                                   int n0, int k0) {
  const int lane = threadIdx.x & 31;
  const int lo = lane & 15, hi = lane >> 4;
  const _Float16* p = base + (size_t)(n0 + lo) * ldk + k0 + 8 * hi;
  FragU f;
  f.u[0] = *(const u32v4*)(p);
  f.u[1] = *(const u32v4*)(p + 16);
  return f.h;
}

// ---------------------------------------------------------------------------
// Kernel 1: fp32 -> fp16 conversion, 4 elements/thread.
// ---------------------------------------------------------------------------
__global__ void cvt_f32_f16(const float* __restrict__ in, _Float16* __restrict__ out,
                            int n4) {
  int i = blockIdx.x * blockDim.x + threadIdx.x;
  if (i < n4) {
    v4f x = *(const v4f*)(in + (size_t)i * 4);
    *(v4h*)(out + (size_t)i * 4) = __builtin_convertvector(x, v4h);
  }
}

// ---------------------------------------------------------------------------
// Kernel 2: C = A[M,K] @ Bw[N,K]^T  (i.e. x @ W.T), WMMA f16->f32.
// Each wave owns a 32x64 C tile: 8 accumulators, 8 WMMAs per 32-wide K step
// against 12 b128 fragment loads; next K-step prefetched into WGP cache.
// MODE 0: out f16 -> [B,H,S,D];  MODE 1: out f16 -> [B,H,D,S] (V transposed);
// MODE 2: out f32 row-major + bias (final projection).
// ---------------------------------------------------------------------------
template <int MODE>
__global__ __launch_bounds__(256) void gemm_wmma(const _Float16* __restrict__ A,
                                                 const _Float16* __restrict__ Bw,
                                                 _Float16* __restrict__ outH,
                                                 float* __restrict__ outF,
                                                 const float* __restrict__ bias,
                                                 int M, int N, int K) {
  const int wave = threadIdx.x >> 5;
  const int wid = blockIdx.x * (blockDim.x >> 5) + wave;
  const int ntiles = N >> 6;                // 64-wide N tiles
  const int mt = wid / ntiles;
  const int nt = wid - mt * ntiles;
  if (mt * 32 >= M) return;  // wave-uniform: EXEC stays all-ones for WMMA
  const int lane = threadIdx.x & 31;
  const int lo = lane & 15, hi = lane >> 4;

  v8f acc[2][4] = {};
  for (int k0 = 0; k0 < K; k0 += 32) {
    // prefetch next K-step's cachelines into WGP cache (global_prefetch_b8)
    if (k0 + 32 < K) {
      __builtin_prefetch(A + (size_t)(mt * 32 + lo + 16 * hi) * K + k0 + 32, 0, 1);
      __builtin_prefetch(Bw + (size_t)(nt * 64 + lo + 16 * hi) * K + k0 + 32, 0, 1);
      __builtin_prefetch(Bw + (size_t)(nt * 64 + 32 + lo + 16 * hi) * K + k0 + 32, 0, 1);
    }
    v16h a0 = load_a_frag(A, K, mt * 32, k0);
    v16h a1 = load_a_frag(A, K, mt * 32 + 16, k0);
#pragma unroll
    for (int tj = 0; tj < 4; ++tj) {
      v16h b = load_b_frag(Bw, K, nt * 64 + tj * 16, k0);
      acc[0][tj] = WMMA_F16(a0, b, acc[0][tj]);
      acc[1][tj] = WMMA_F16(a1, b, acc[1][tj]);
    }
  }

#pragma unroll
  for (int ti = 0; ti < 2; ++ti)
#pragma unroll
    for (int tj = 0; tj < 4; ++tj)
#pragma unroll
      for (int r = 0; r < 8; ++r) {
        const int i = mt * 32 + ti * 16 + r + 8 * hi;  // C row  (M index)
        const int j = nt * 64 + tj * 16 + lo;          // C col  (N index)
        const float v = acc[ti][tj][r];
        if (MODE == 0) {
          const int b = i >> 11, s = i & (SEQ - 1);
          const int h = j >> 6, d = j & (HD - 1);
          outH[((size_t)(b * NH + h) * SEQ + s) * HD + d] = (_Float16)v;
        } else if (MODE == 1) {
          const int b = i >> 11, s = i & (SEQ - 1);
          const int h = j >> 6, d = j & (HD - 1);
          outH[((size_t)(b * NH + h) * HD + d) * SEQ + s] = (_Float16)v;
        } else {
          outF[(size_t)i * N + j] = v + bias[j];
        }
      }
}

// ---------------------------------------------------------------------------
// Kernel 3: causal flash attention. One wave per (b,h, 16-row query tile).
// Streams 32-wide key blocks; scores never touch HBM (reg/LDS only).
// ---------------------------------------------------------------------------
__global__ __launch_bounds__(128) void flash_attn(const _Float16* __restrict__ Qp,
                                                  const _Float16* __restrict__ Kp,
                                                  const _Float16* __restrict__ Vt,
                                                  _Float16* __restrict__ AO) {
  __shared__ alignas(16) _Float16 ldsP[4][16 * 32];  // P tile per wave (1 KB)
  const int wave = threadIdx.x >> 5;
  const int wid = blockIdx.x * 4 + wave;
  const int qt = wid & (SEQ / 16 - 1);  // query tile 0..127
  const int bh = wid >> 7;              // 0..31 (b*16 + h)
  const int lane = threadIdx.x & 31;
  const int lo = lane & 15, hi = lane >> 4;
  const int s0 = qt * 16;

  const _Float16* Qb = Qp + (size_t)bh * SEQ * HD;  // [s][d]
  const _Float16* Kb = Kp + (size_t)bh * SEQ * HD;  // [t][d] == (K^T)^T
  const _Float16* Vb = Vt + (size_t)bh * HD * SEQ;  // [d][t] == V^T

  // Q tile fragments (rows s0..s0+15, d 0..31 and 32..63) kept in registers.
  const v16h aQ0 = load_a_frag(Qb, HD, s0, 0);
  const v16h aQ1 = load_a_frag(Qb, HD, s0, 32);

  float mrow[8], lrow[8];
  v8f o[4] = {};
#pragma unroll
  for (int r = 0; r < 8; ++r) {
    mrow[r] = -__builtin_inff();
    lrow[r] = 0.0f;
  }

  const int nblocks = (s0 + 47) >> 5;  // ceil((s0+16)/32): causal truncation
  for (int blk = 0; blk < nblocks; ++blk) {
    const int t0 = blk * 32;

    // prefetch next key block's K rows into WGP cache
    if (blk + 1 < nblocks)
      __builtin_prefetch(Kb + (size_t)(t0 + 32 + lane) * HD, 0, 1);

    // scores: S = Q (16x64) @ K^T (64x32) -> two 16x16 tiles, 4 WMMAs
    v8f c0 = {}, c1 = {};
    c0 = WMMA_F16(aQ0, load_b_frag(Kb, HD, t0, 0), c0);
    c0 = WMMA_F16(aQ1, load_b_frag(Kb, HD, t0, 32), c0);
    c1 = WMMA_F16(aQ0, load_b_frag(Kb, HD, t0 + 16, 0), c1);
    c1 = WMMA_F16(aQ1, load_b_frag(Kb, HD, t0 + 16, 32), c1);

    // online softmax over rows (row spread across 16 lanes of this half-wave)
#pragma unroll
    for (int r = 0; r < 8; ++r) {
      const int s = s0 + r + 8 * hi;
      float x0 = c0[r] * 0.125f;  // 1/sqrt(64)
      float x1 = c1[r] * 0.125f;
      if (t0 + lo > s) x0 = -__builtin_inff();       // causal mask
      if (t0 + 16 + lo > s) x1 = -__builtin_inff();

      float mx = fmaxf(x0, x1);
      for (int off = 8; off > 0; off >>= 1) mx = fmaxf(mx, __shfl_xor(mx, off, 16));
      const float newm = fmaxf(mrow[r], mx);
      const float alpha = __expf(mrow[r] - newm);
      const float e0 = __expf(x0 - newm);
      const float e1 = __expf(x1 - newm);
      float rs = e0 + e1;
      for (int off = 8; off > 0; off >>= 1) rs += __shfl_xor(rs, off, 16);
      lrow[r] = lrow[r] * alpha + rs;
      mrow[r] = newm;
#pragma unroll
      for (int t = 0; t < 4; ++t) o[t][r] *= alpha;

      // transpose P through LDS: C layout (row on VGPR) -> A layout (row on lane)
      ldsP[wave][(r + 8 * hi) * 32 + lo] = (_Float16)e0;
      ldsP[wave][(r + 8 * hi) * 32 + 16 + lo] = (_Float16)e1;
    }

    // O += P (16x32) @ V (32x64): B frags from V^T rows are contiguous
    const v16h aP = load_a_frag(&ldsP[wave][0], 32, 0, 0);
#pragma unroll
    for (int t = 0; t < 4; ++t)
      o[t] = WMMA_F16(aP, load_b_frag(Vb, SEQ, t * 16, t0), o[t]);
  }

  // normalize and write f16 attention output in [B][S][E] (E = h*64+d)
  const int b = bh >> 4, h = bh & 15;
#pragma unroll
  for (int r = 0; r < 8; ++r) {
    const float inv = 1.0f / lrow[r];
    const int s = s0 + r + 8 * hi;
    const size_t row = ((size_t)b * SEQ + s) * EMB + h * HD;
#pragma unroll
    for (int t = 0; t < 4; ++t)
      AO[row + t * 16 + lo] = (_Float16)(o[t][r] * inv);
  }
}

// ---------------------------------------------------------------------------
// Host-side launcher
// ---------------------------------------------------------------------------
extern "C" void kernel_launch(void* const* d_in, const int* in_sizes, int n_in,
                              void* d_out, int out_size, void* d_ws, size_t ws_size,
                              hipStream_t stream) {
  const float* q = (const float*)d_in[0];
  const float* k = (const float*)d_in[1];
  const float* v = (const float*)d_in[2];
  const float* Wq = (const float*)d_in[3];
  const float* Wk = (const float*)d_in[4];
  const float* Wv = (const float*)d_in[5];
  const float* Wo = (const float*)d_in[6];
  const float* bo = (const float*)d_in[7];
  float* out = (float*)d_out;

  const int NTOK = BATCH * SEQ;       // 4096 rows
  const int NX = NTOK * EMB;          // 4,194,304 elements
  const int NW = EMB * EMB;           // 1,048,576 elements

  _Float16* ws = (_Float16*)d_ws;
  _Float16* qh = ws;                  // f16 inputs
  _Float16* kh = qh + NX;
  _Float16* vh = kh + NX;
  _Float16* WqH = vh + NX;            // f16 weights [N][K] = B^T layout
  _Float16* WkH = WqH + NW;
  _Float16* WvH = WkH + NW;
  _Float16* WoH = WvH + NW;
  _Float16* Qp = WoH + NW;            // [B,H,S,D]
  _Float16* Kp = Qp + NX;             // [B,H,S,D]
  _Float16* Vt = Kp + NX;             // [B,H,D,S]  (pre-transposed V)
  _Float16* AO = Vt + NX;             // [B,S,E]    (attention output, f16)

  // 1) fp32 -> fp16
  cvt_f32_f16<<<NX / 4 / 256, 256, 0, stream>>>(q, qh, NX / 4);
  cvt_f32_f16<<<NX / 4 / 256, 256, 0, stream>>>(k, kh, NX / 4);
  cvt_f32_f16<<<NX / 4 / 256, 256, 0, stream>>>(v, vh, NX / 4);
  cvt_f32_f16<<<NW / 4 / 256, 256, 0, stream>>>(Wq, WqH, NW / 4);
  cvt_f32_f16<<<NW / 4 / 256, 256, 0, stream>>>(Wk, WkH, NW / 4);
  cvt_f32_f16<<<NW / 4 / 256, 256, 0, stream>>>(Wv, WvH, NW / 4);
  cvt_f32_f16<<<NW / 4 / 256, 256, 0, stream>>>(Wo, WoH, NW / 4);

  // 2) projections: (4096/32)*(1024/64) = 2048 wave-tiles, 8 waves/block
  const int gemm_blocks = (NTOK / 32) * (EMB / 64) / 8;  // 256
  gemm_wmma<0><<<gemm_blocks, 256, 0, stream>>>(qh, WqH, Qp, nullptr, nullptr,
                                                NTOK, EMB, EMB);
  gemm_wmma<0><<<gemm_blocks, 256, 0, stream>>>(kh, WkH, Kp, nullptr, nullptr,
                                                NTOK, EMB, EMB);
  gemm_wmma<1><<<gemm_blocks, 256, 0, stream>>>(vh, WvH, Vt, nullptr, nullptr,
                                                NTOK, EMB, EMB);

  // 3) flash attention: 32 (b,h) * 128 query tiles = 4096 waves, 4 waves/block
  flash_attn<<<BATCH * NH * (SEQ / 16) / 4, 128, 0, stream>>>(Qp, Kp, Vt, AO);

  // 4) output projection + bias -> fp32
  gemm_wmma<2><<<gemm_blocks, 256, 0, stream>>>(AO, WoH, nullptr, out, bo,
                                                NTOK, EMB, EMB);
}
